// BinaryGRUModelModify_6442450944379
// MI455X (gfx1250) — matrix operand order repaired and checked by reading, not compiled, not run on hardware
//
#include <hip/hip_runtime.h>
#include <hip/hip_bf16.h>
#include <stdint.h>

// ---------------------------------------------------------------------------
// Stacked 2-layer GRU forward loss for MI455X (gfx1250), wave32 + WMMA bf16.
//
//   * Weights converted once per launch to bf16, transposed to [n_out][k] rows
//     so B-matrix WMMA fragments are two contiguous b128 LDS loads per lane.
//   * Persistent kernel: 16 workgroups, each owns a 16-column slice of H=256.
//     All 12 per-step [64,256]x[256,256] matmuls run as v_wmma_f32_16x16x32_bf16.
//   * Weight slices staged into LDS by the Tensor Data Mover (tensor_load_to_lds
//     with pad_enable giving the bank-conflict-avoiding 528B row stride),
//     waited with s_wait_tensorcnt. Fallback: manual b128 staging.
//   * Hidden state / r*S1 / layer-1 output broadcast through L2 as bf16.
//   * Grid sync: monotonic global atomic barrier (16 WGs always co-resident).
// ---------------------------------------------------------------------------

#define LAYERS   2
#define BATCH    64
#define TMAX     512
#define DIM      256
#define HID      256
#define NWG      16
#define NTHREADS 256
#define SLICE    16           // output columns owned per workgroup
#define WROW     264          // padded LDS row (halves): kills 512B bank alias

typedef __attribute__((ext_vector_type(16))) __bf16 v16bf;
typedef __attribute__((ext_vector_type(8)))  float  v8f;
typedef __attribute__((ext_vector_type(4)))  unsigned int v4u;
typedef __attribute__((ext_vector_type(8)))  int    v8i;
typedef __attribute__((ext_vector_type(4)))  int    v4i;

#if __has_builtin(__builtin_amdgcn_tensor_load_to_lds)
#define USE_TDM 1
#else
#define USE_TDM 0
#endif

union Frag {
  v16bf v;
  uint4 q[2];
};

__device__ __forceinline__ unsigned short f2bf(float f) {
  unsigned u = __float_as_uint(f);
  u += 0x7FFFu + ((u >> 16) & 1u);      // round-to-nearest-even
  return (unsigned short)(u >> 16);
}

__device__ __forceinline__ void grid_barrier(unsigned* bar, unsigned expected) {
  __syncthreads();
  if (threadIdx.x == 0) {
    __threadfence();
    atomicAdd(bar, 1u);
    while (*((volatile unsigned*)bar) < expected) {
      __builtin_amdgcn_s_sleep(1);
    }
    __threadfence();
  }
  __syncthreads();
}

// One 16x16 output tile: C += A[16x256] * B[256x16] via 8 x wmma_f32_16x16x32_bf16.
// A: global bf16, row-major [.,256]. B: LDS, transposed rows [16][WROW].
__device__ __forceinline__ void mm_tile(const unsigned short* __restrict__ Ag,
                                        const unsigned short* lwp,   // LDS [16][WROW]
                                        float* accp,                 // LDS [64][16]
                                        int mt, int lane) {
  const int g    = lane >> 4;
  const int m    = lane & 15;
  const int row  = mt * 16 + m;
  const unsigned short* arow = Ag + row * HID;
  const unsigned short* brow = lwp + m * WROW;   // lane's N column = lane&15
  v8f c = {0.f, 0.f, 0.f, 0.f, 0.f, 0.f, 0.f, 0.f};
#pragma unroll
  for (int kk = 0; kk < HID; kk += 32) {
    Frag a, b;
    // A layout (ISA 7.12.2, 16-bit 16x32): halves 0..7 -> K=kk+8g.., 8..15 -> K=kk+16+8g..
    a.q[0] = *(const uint4*)(arow + kk + 8 * g);
    a.q[1] = *(const uint4*)(arow + kk + 16 + 8 * g);
    // B layout: same as A: halves 0..7 -> K=kk+8g.., 8..15 -> K=kk+16+8g..
    b.q[0] = *(const uint4*)(brow + kk + 8 * g);
    b.q[1] = *(const uint4*)(brow + kk + 16 + 8 * g);
    c = __builtin_amdgcn_wmma_f32_16x16x32_bf16(false, a.v, false, b.v,
                                                (short)0, c, false, false);
  }
#pragma unroll
  for (int v = 0; v < 8; v++)                    // C: lane<16 M=v, lane>=16 M=v+8
    accp[(mt * 16 + v + 8 * g) * SLICE + m] = c[v];
}

#if USE_TDM
// Tensor Data Mover: DMA one transposed weight slice (16 rows x 256 bf16) into
// LDS, inserting 4 DWORDs of padding every 128 DWORDs (row stride 264 halves).
__device__ __forceinline__ void tdm_stage(const unsigned short* __restrict__ WT,
                                          unsigned short* lds_dst,
                                          int m, int nbase) {
  typedef __attribute__((address_space(3))) unsigned short lds_us;
  unsigned lds_off = (unsigned)(unsigned long long)(lds_us*)lds_dst;
  unsigned long long gaddr =
      (unsigned long long)(const void*)(WT + (size_t)m * 65536 + (size_t)nbase * 256);

  // D# group 0: count=1 | lds_addr | global_addr[56:0] | type=2
  v4u g0;
  g0.x = 1u;
  g0.y = lds_off;
  g0.z = (unsigned)(gaddr & 0xFFFFFFFFu);
  g0.w = (unsigned)((gaddr >> 32) & 0x1FFFFFFu) | (2u << 30);

  // D# group 1: data_size=2B, pad_enable, pad_interval=128dw, pad_amount=4dw,
  // tensor 256x256, tile 256x16, dim0_stride=256, dim1_stride=65536.
  v8i g1;
  g1[0] = (1 << 16) | (1 << 20) | (6 << 22) | (3 << 25);
  g1[1] = (int)((256u & 0xFFFFu) << 16);        // tensor_dim0[15:0]
  g1[2] = (int)((256u & 0xFFFFu) << 16);        // dim0[31:16]=0 | tensor_dim1[15:0]
  g1[3] = (int)(256u << 16);                    // dim1[31:16]=0 | tile_dim0=256
  g1[4] = 16;                                   // tile_dim1=16, tile_dim2=0
  g1[5] = 256;                                  // tensor_dim0_stride[31:0]
  g1[6] = 0;                                    // stride0[47:32]=0 | stride1[15:0]=0
  g1[7] = 1;                                    // tensor_dim1_stride[47:16] = 1 (65536)

  v4i g2 = {0, 0, 0, 0};                        // 2-D tile: dims 2..4 unused
  v4i g3 = {0, 0, 0, 0};

#if defined(__clang_major__) && (__clang_major__ >= 23)
  v8i gx = {0, 0, 0, 0, 0, 0, 0, 0};
  __builtin_amdgcn_tensor_load_to_lds(g0, g1, g2, g3, gx, 0);
#else
  __builtin_amdgcn_tensor_load_to_lds(g0, g1, g2, g3, 0);
#endif
}
#endif

// Stage `nmat` transposed weight slices (rows nbase..nbase+15, 256 K each) to LDS.
__device__ __forceinline__ void stage_weights(const unsigned short* __restrict__ WT,
                                              unsigned short* lds_w,
                                              int mstart, int nmat, int nbase, int tid) {
#if USE_TDM
  if ((tid >> 5) == 0) {                        // one wave issues the DMAs
#pragma unroll
    for (int p = 0; p < nmat; ++p)
      tdm_stage(WT, lds_w + p * SLICE * WROW, mstart + p, nbase);
    __builtin_amdgcn_s_wait_tensorcnt(0);
  }
#else
  const int total = nmat * SLICE * 32;          // uint4 chunks
  for (int idx = tid; idx < total; idx += NTHREADS) {
    int p   = idx >> 9;
    int rem = idx & 511;
    int n   = rem >> 5;
    int q   = rem & 31;
    uint4 src = *(const uint4*)(WT + (size_t)(mstart + p) * 65536
                                   + (size_t)(nbase + n) * 256 + q * 8);
    *(uint4*)(lds_w + (p * SLICE + n) * WROW + q * 8) = src;
  }
#endif
}

__global__ void __launch_bounds__(NTHREADS, 1)
gru_persistent(const float* __restrict__ x_data,
               const int*   __restrict__ x_length,
               const float* __restrict__ x_label,
               const float* __restrict__ Wo,
               const unsigned short* __restrict__ WT,
               float* __restrict__ S1f,
               float* __restrict__ S2f,
               unsigned short* __restrict__ S1b,
               unsigned short* __restrict__ xb,
               unsigned short* __restrict__ rsb,
               unsigned* __restrict__ bar,
               float* __restrict__ out)
{
  __shared__ unsigned short lds_w[4 * SLICE * WROW];   // 33 KB weight slices
  __shared__ float lds_acc[4 * BATCH * SLICE];         // 16 KB f32 accum tiles
  __shared__ float lds_z[BATCH * SLICE];               // 4 KB z gate slice

  const int tid   = threadIdx.x;
  const int wg    = blockIdx.x;
  const int nbase = wg * SLICE;
  const int w     = tid >> 5;
  const int lane  = tid & 31;

  unsigned bseq   = 0;
  float lossAcc   = 0.f;

  for (int t = 0; t < TMAX; ++t) {
    // ---- phase 0: convert x[:, t, slice] f32 -> bf16 broadcast buffer ----
    for (int e = tid; e < BATCH * SLICE; e += NTHREADS) {
      int b = e >> 4, c = e & 15, gc = nbase + c;
      xb[b * DIM + gc] = f2bf(x_data[((size_t)b * TMAX + t) * DIM + gc]);
    }
    grid_barrier(bar, NWG * (++bseq));

    for (int l = 0; l < LAYERS; ++l) {
      const unsigned short* inpb = (l == 0) ? xb : S1b;       // layer1 in = hn of layer0
      const unsigned short* hb   = S1b + l * BATCH * HID;     // previous hidden (bf16)
      float*          S1l  = S1f + l * BATCH * HID;
      float*          S2l  = S2f + l * BATCH * HID;
      unsigned short* S1bl = S1b + l * BATCH * HID;

      // -------- phase A: z = sig(in@Wz + S1@Uz), r = sig(in@Wr + S1@Ur) --------
      stage_weights(WT, lds_w, l * 6 + 0, 4, nbase, tid);
      { // prefetch phase-B weight slices (Wh, Uh) toward L2/WGP$
        const unsigned short* pf = WT + (size_t)(l * 6 + 4) * 65536 + (size_t)nbase * 256;
        __builtin_prefetch(pf + tid * 32, 0, 0);
        __builtin_prefetch(pf + 65536 + tid * 32, 0, 0);
      }
      __syncthreads();
#pragma unroll
      for (int j = 2 * w; j < 2 * w + 2; ++j) {   // 16 tile-jobs over 8 waves
        int p  = j >> 2;                          // 0:Wz 1:Uz 2:Wr 3:Ur
        int mt = j & 3;
        const unsigned short* Ag = (p & 1) ? hb : inpb;
        mm_tile(Ag, lds_w + p * SLICE * WROW, lds_acc + p * BATCH * SLICE, mt, lane);
      }
      __syncthreads();
      for (int e = tid; e < BATCH * SLICE; e += NTHREADS) {
        int b = e >> 4, c = e & 15, gc = nbase + c;
        float az = lds_acc[e] + lds_acc[BATCH * SLICE + e];
        float ar = lds_acc[2 * BATCH * SLICE + e] + lds_acc[3 * BATCH * SLICE + e];
        float z  = 1.f / (1.f + __expf(-az));
        float r  = 1.f / (1.f + __expf(-ar));
        lds_z[e] = z;
        rsb[b * HID + gc] = f2bf(r * S1l[b * HID + gc]);
      }
      grid_barrier(bar, NWG * (++bseq));

      // -------- phase B: h = tanh(in@Wh + (r*S1)@Uh), state update --------
      stage_weights(WT, lds_w, l * 6 + 4, 2, nbase, tid);
      __syncthreads();
      {
        int p  = w >> 2;                          // 0:Wh 1:Uh
        int mt = w & 3;
        const unsigned short* Ag = (p == 0) ? inpb : rsb;
        mm_tile(Ag, lds_w + p * SLICE * WROW, lds_acc + p * BATCH * SLICE, mt, lane);
      }
      __syncthreads();
      for (int e = tid; e < BATCH * SLICE; e += NTHREADS) {
        int b = e >> 4, c = e & 15, gc = nbase + c;
        float h   = tanhf(lds_acc[e] + lds_acc[BATCH * SLICE + e]);
        float z   = lds_z[e];
        float s1  = S1l[b * HID + gc];
        float s2  = S2l[b * HID + gc];
        float hn  = (1.f - z) * s1 + z * h;
        float s2n = z * s2 + (1.f - z) * h;
        S1l[b * HID + gc]  = hn;
        S2l[b * HID + gc]  = s2n;
        S1bl[b * HID + gc] = f2bf(hn);
      }
      grid_barrier(bar, NWG * (++bseq));
    }

    // ---- scoring + masked loss (WG0, overlapped with others' next phase 0) ----
    if (wg == 0 && tid < BATCH) {
      const float* h2 = S1f + BATCH * HID + tid * HID;   // layer-1 hidden (f32)
      float dot = 0.f;
#pragma unroll 8
      for (int d = 0; d < HID; ++d) dot += h2[d] * Wo[d * 2 + 1];
      float sc = 1.f / (1.f + __expf(-dot));
      if (t < x_length[tid]) {
        float dlt = x_label[tid] - sc;
        lossAcc += dlt * dlt;
      }
    }
  }

  if (wg == 0 && tid < BATCH) atomicAdd(out, lossAcc);
}

// --- prep kernels -----------------------------------------------------------

__global__ void convert_weights(const float* __restrict__ Wz, const float* __restrict__ Uz,
                                const float* __restrict__ Wr, const float* __restrict__ Ur,
                                const float* __restrict__ Wh, const float* __restrict__ Uh,
                                unsigned short* __restrict__ WT) {
  size_t i = (size_t)blockIdx.x * blockDim.x + threadIdx.x;
  if (i >= (size_t)12 * 65536) return;
  int m = (int)(i >> 16);
  int rem = (int)(i & 65535);
  int n = rem >> 8;           // output column
  int k = rem & 255;          // contraction index
  int l = m / 6, gate = m % 6;
  const float* src;
  switch (gate) {
    case 0: src = Wz; break; case 1: src = Uz; break; case 2: src = Wr; break;
    case 3: src = Ur; break; case 4: src = Wh; break; default: src = Uh;
  }
  float v = src[(size_t)l * 65536 + (size_t)k * 256 + n];
  WT[(size_t)m * 65536 + (size_t)n * 256 + k] = f2bf(v);   // transposed bf16
}

__global__ void init_state(float* __restrict__ S1f, unsigned* __restrict__ bar,
                           float* __restrict__ out) {
  size_t i = (size_t)blockIdx.x * blockDim.x + threadIdx.x;
  unsigned* z = (unsigned*)S1f;              // S1f|S2f|S1b contiguous: 327680 B
  if (i < 81920) z[i] = 0u;
  if (i == 0) { *bar = 0u; *out = 0.f; }
}

// --- launch -----------------------------------------------------------------

extern "C" void kernel_launch(void* const* d_in, const int* in_sizes, int n_in,
                              void* d_out, int out_size, void* d_ws, size_t ws_size,
                              hipStream_t stream) {
  const float* x_data   = (const float*)d_in[0];
  const int*   x_length = (const int*)d_in[1];
  const float* x_label  = (const float*)d_in[2];
  const float* Wz = (const float*)d_in[3];
  const float* Uz = (const float*)d_in[4];
  const float* Wr = (const float*)d_in[5];
  const float* Ur = (const float*)d_in[6];
  const float* Wh = (const float*)d_in[7];
  const float* Uh = (const float*)d_in[8];
  const float* Wo = (const float*)d_in[9];

  char* ws = (char*)d_ws;
  unsigned short* WT  = (unsigned short*)(ws);                // 1,572,864 B
  float*          S1f = (float*)(ws + 1572864);               //   131,072 B
  float*          S2f = (float*)(ws + 1703936);               //   131,072 B
  unsigned short* S1b = (unsigned short*)(ws + 1835008);      //    65,536 B
  unsigned short* xb  = (unsigned short*)(ws + 1900544);      //    32,768 B
  unsigned short* rsb = (unsigned short*)(ws + 1933312);      //    32,768 B
  unsigned*       bar = (unsigned*)(ws + 1966080);
  float*          out = (float*)d_out;

  hipLaunchKernelGGL(convert_weights, dim3(3072), dim3(256), 0, stream,
                     Wz, Uz, Wr, Ur, Wh, Uh, WT);
  hipLaunchKernelGGL(init_state, dim3(320), dim3(256), 0, stream, S1f, bar, out);
  hipLaunchKernelGGL(gru_persistent, dim3(NWG), dim3(NTHREADS), 0, stream,
                     x_data, x_length, x_label, Wo, WT, S1f, S2f, S1b, xb, rsb,
                     bar, out);
}
